// FashionMNIST_Type1_Template_Unroll_74217034875446
// MI455X (gfx1250) — matrix operand order repaired and checked by reading, not compiled, not run on hardware
//
#include <hip/hip_runtime.h>

typedef _Float16 half_t;
typedef __attribute__((ext_vector_type(16))) _Float16 v16h;
typedef __attribute__((ext_vector_type(8)))  _Float16 v8h;
typedef __attribute__((ext_vector_type(8)))  float    v8f;

#define B_BATCH 8192
#define K_FC1   2304
#define N_FC1   1024
#define BROW_PAD 40   // B-tile LDS row stride in halfs (bank-conflict-free, 16B aligned)

static __device__ __forceinline__ v16h load_frag(const half_t* p0, const half_t* p1) {
    v8h lo = *(const v8h*)p0;
    v8h hi = *(const v8h*)p1;
    return __builtin_shufflevector(lo, hi, 0,1,2,3,4,5,6,7,8,9,10,11,12,13,14,15);
}

// ---------------------------------------------------------------------------
// prep_weights: fc1w f32 -> f16 (row-major 1024x2304), and conv2w
// (64c, 36p, 48k) -> w2t (36p, 64c, 64k) f16, zero-padded k 48->64.
// ---------------------------------------------------------------------------
__global__ void prep_weights(const float* __restrict__ fc1w,
                             const float* __restrict__ conv2w,
                             half_t* __restrict__ w1h,
                             half_t* __restrict__ w2t)
{
    const int N1 = N_FC1 * K_FC1;          // 2359296
    const int N2 = 36 * 64 * 64;           // 147456
    for (int i = blockIdx.x * blockDim.x + threadIdx.x; i < N1 + N2;
         i += gridDim.x * blockDim.x) {
        if (i < N1) {
            w1h[i] = (half_t)fc1w[i];
        } else {
            int j = i - N1;
            int k = j & 63;
            int c = (j >> 6) & 63;
            int p = j >> 12;               // 0..35
            float v = 0.0f;
            if (k < 48) v = conv2w[(c * 36 + p) * 48 + k];
            w2t[j] = (half_t)v;
        }
    }
}

// ---------------------------------------------------------------------------
// build_p2: fused conv1 (locally-connected 2x2 stride2, relu) + patch2 gather.
// Writes a2[b][p=py*6+px][k], k = c*16 + dy*4 + dx (k>=48 zero pad), f16.
// ---------------------------------------------------------------------------
__global__ void build_p2(const float* __restrict__ x,     // B x 784
                         const float* __restrict__ w1,    // 3 x 196 x 4
                         half_t* __restrict__ a2)         // B x 36 x 64
{
    long long i = (long long)blockIdx.x * blockDim.x + threadIdx.x;
    const long long total = (long long)B_BATCH * 36 * 64;
    if (i >= total) return;
    int k = (int)(i & 63);
    int p = (int)((i >> 6) % 36);
    long long b = i / (36 * 64);
    float v = 0.0f;
    if (k < 48) {
        int c  = k >> 4;
        int dy = (k >> 2) & 3;
        int dx = k & 3;
        int py = p / 6, px = p % 6;
        int row = 2 * py + dy;             // 0..13
        int col = 2 * px + dx;             // 0..13
        const float* xb = x + b * 784;
        const float* wp = w1 + (c * 196 + row * 14 + col) * 4;
        int y0 = 2 * row, x0 = 2 * col;
        float s = xb[y0 * 28 + x0]           * wp[0]
                + xb[y0 * 28 + x0 + 1]       * wp[1]
                + xb[(y0 + 1) * 28 + x0]     * wp[2]
                + xb[(y0 + 1) * 28 + x0 + 1] * wp[3];
        v = s > 0.0f ? s : 0.0f;
    }
    a2[i] = (half_t)v;
}

// ---------------------------------------------------------------------------
// conv2_wmma: per position p, 32(b) x 32(c) register tile per wave,
// 2x2 wmma tiles x 2 k-steps. out2[b][c*36+p] = relu, f16.
// ---------------------------------------------------------------------------
__global__ void conv2_wmma(const half_t* __restrict__ a2,   // B x 36 x 64
                           const half_t* __restrict__ w2t,  // 36 x 64(c) x 64(k)
                           half_t* __restrict__ out2)       // B x 2304
{
    int wave = blockIdx.x * (blockDim.x >> 5) + (threadIdx.x >> 5);
    int lane = threadIdx.x & 31;
    int ct = wave & 1;                  // 2 c-tiles of 32
    int p  = (wave >> 1) % 36;
    int bt = wave / (2 * 36);           // 256 b-tiles of 32
    int b0 = bt * 32;
    int c0 = ct * 32;
    int ln = lane & 15;
    int lh = lane >> 4;

    const half_t* arow[2];
    const half_t* brow[2];
#pragma unroll
    for (int s = 0; s < 2; ++s)
        arow[s] = a2 + ((long long)(b0 + s * 16 + ln) * 36 + p) * 64;
#pragma unroll
    for (int t = 0; t < 2; ++t)
        brow[t] = w2t + ((long long)(p * 64) + (c0 + t * 16 + ln)) * 64;

    v8f acc[2][2] = {};
#pragma unroll
    for (int ko = 0; ko < 64; ko += 32) {
        v16h a[2], bm[2];
#pragma unroll
        for (int s = 0; s < 2; ++s)
            a[s] = load_frag(arow[s] + ko + lh * 16, arow[s] + ko + lh * 16 + 8);
#pragma unroll
        for (int t = 0; t < 2; ++t)
            bm[t] = load_frag(brow[t] + ko + lh * 16, brow[t] + ko + lh * 16 + 8);
#pragma unroll
        for (int s = 0; s < 2; ++s)
#pragma unroll
            for (int t = 0; t < 2; ++t)
                acc[s][t] = __builtin_amdgcn_wmma_f32_16x16x32_f16(
                    false, a[s], false, bm[t], (short)0, acc[s][t], false, false);
    }
#pragma unroll
    for (int s = 0; s < 2; ++s)
#pragma unroll
        for (int t = 0; t < 2; ++t) {
            int c = c0 + t * 16 + ln;
#pragma unroll
            for (int r = 0; r < 8; ++r) {
                int b = b0 + s * 16 + r + 8 * lh;
                float v = acc[s][t][r];
                out2[(long long)b * K_FC1 + c * 36 + p] = (half_t)(v > 0.0f ? v : 0.0f);
            }
        }
}

// ---------------------------------------------------------------------------
// fc1_wmma: (8192 x 2304) @ (2304 x 1024)^T.
// Block tile 256(b) x 64(n): 8 waves share the B tile via double-buffered LDS
// (row stride 40 halfs -> conflict-free b128 fragment reads). Each wave:
// 32(b) x 64(n) = 2x4 WMMA tiles. A streams from global with prefetch.
// h = relu(out2 @ fc1w^T), stored f16.
// ---------------------------------------------------------------------------
__global__ void __launch_bounds__(256) fc1_wmma(
                         const half_t* __restrict__ a1,   // B x 2304
                         const half_t* __restrict__ w1h,  // 1024 x 2304
                         half_t* __restrict__ h)          // B x 1024
{
    __shared__ __align__(16) half_t sB[2][64 * BROW_PAD];   // 2 x 5 KB

    int tid  = threadIdx.x;
    int wave = tid >> 5;
    int lane = tid & 31;
    int nblk = blockIdx.x & 15;         // 16 n-blocks of 64
    int bblk = blockIdx.x >> 4;         // 32 b-blocks of 256
    int n0 = nblk * 64;
    int b0 = bblk * 256 + wave * 32;
    int ln = lane & 15, lh = lane >> 4;

    // B copy mapping: thread -> (row 0..63, 16B segment 0..3), one b128 each.
    int crow = tid >> 2;
    int cseg = tid & 3;
    const half_t* csrc = w1h + (long long)(n0 + crow) * K_FC1 + cseg * 8;
    half_t* cdst[2];
    cdst[0] = &sB[0][crow * BROW_PAD + cseg * 8];
    cdst[1] = &sB[1][crow * BROW_PAD + cseg * 8];

    const half_t* arow[2];
#pragma unroll
    for (int s = 0; s < 2; ++s)
        arow[s] = a1 + (long long)(b0 + s * 16 + ln) * K_FC1;

    // prologue: stage B(k=0)
    *(v8h*)cdst[0] = *(const v8h*)csrc;
    __syncthreads();

    v8f acc[2][4] = {};
    for (int ko = 0; ko < K_FC1; ko += 32) {
        int cur = (ko >> 5) & 1;
        bool notlast = (ko + 32 < K_FC1);

        // next B tile: global -> regs, overlapped with this step's WMMAs
        v8h cnext;
        if (notlast) cnext = *(const v8h*)(csrc + ko + 32);

        // prefetch A stream 8 k-steps (512B) ahead
        if (ko + 256 < K_FC1) {
#pragma unroll
            for (int s = 0; s < 2; ++s) __builtin_prefetch(arow[s] + ko + 256, 0, 3);
        }

        v16h a[2];
#pragma unroll
        for (int s = 0; s < 2; ++s)
            a[s] = load_frag(arow[s] + ko + lh * 16, arow[s] + ko + lh * 16 + 8);

        v16h bm[4];
#pragma unroll
        for (int t = 0; t < 4; ++t) {
            const half_t* bp = &sB[cur][(t * 16 + ln) * BROW_PAD + lh * 16];
            bm[t] = load_frag(bp, bp + 8);
        }

#pragma unroll
        for (int s = 0; s < 2; ++s)
#pragma unroll
            for (int t = 0; t < 4; ++t)
                acc[s][t] = __builtin_amdgcn_wmma_f32_16x16x32_f16(
                    false, a[s], false, bm[t], (short)0, acc[s][t], false, false);

        if (notlast) {
            __syncthreads();                 // all waves done reading buf[next]
            *(v8h*)cdst[cur ^ 1] = cnext;    // ds_store_b128
            __syncthreads();                 // buf[next] ready
        }
    }

#pragma unroll
    for (int s = 0; s < 2; ++s)
#pragma unroll
        for (int t = 0; t < 4; ++t) {
            int n = n0 + t * 16 + ln;
#pragma unroll
            for (int r = 0; r < 8; ++r) {
                int b = b0 + s * 16 + r + 8 * lh;
                float v = acc[s][t][r];
                h[(long long)b * N_FC1 + n] = (half_t)(v > 0.0f ? v : 0.0f);
            }
        }
}

// ---------------------------------------------------------------------------
// fc2: out[b][j] = sum_n h[b][n] * fc2w[j][n], j < 10. f32 accumulate.
// ---------------------------------------------------------------------------
__global__ void fc2_kernel(const half_t* __restrict__ h,    // B x 1024
                           const float* __restrict__ fc2w,  // 10 x 1024
                           float* __restrict__ out)         // B x 10
{
    int tid = blockIdx.x * blockDim.x + threadIdx.x;
    int j = tid & 15;
    int b = tid >> 4;
    if (b >= B_BATCH || j >= 10) return;
    const half_t* hr = h + (long long)b * N_FC1;
    const float*  wr = fc2w + j * N_FC1;
    float acc = 0.0f;
#pragma unroll 4
    for (int k = 0; k < N_FC1; k += 8) {
        v8h hv = *(const v8h*)(hr + k);
        float4 w0 = *(const float4*)(wr + k);
        float4 w1 = *(const float4*)(wr + k + 4);
        acc += (float)hv[0] * w0.x + (float)hv[1] * w0.y
             + (float)hv[2] * w0.z + (float)hv[3] * w0.w
             + (float)hv[4] * w1.x + (float)hv[5] * w1.y
             + (float)hv[6] * w1.z + (float)hv[7] * w1.w;
    }
    out[b * 10 + j] = acc;
}

// ---------------------------------------------------------------------------
extern "C" void kernel_launch(void* const* d_in, const int* in_sizes, int n_in,
                              void* d_out, int out_size, void* d_ws, size_t ws_size,
                              hipStream_t stream) {
    const float* x      = (const float*)d_in[0];   // 8192 x 1 x 28 x 28
    const float* conv1w = (const float*)d_in[1];   // 3 x 196 x 1 x 2 x 2
    const float* conv2w = (const float*)d_in[2];   // 64 x 36 x 3 x 4 x 4
    const float* fc1w   = (const float*)d_in[3];   // 1024 x 2304
    const float* fc2w   = (const float*)d_in[4];   // 10 x 1024
    float* out = (float*)d_out;

    char* ws = (char*)d_ws;
    size_t off = 0;
    auto wsalloc = [&](size_t bytes) {
        void* p = ws + off;
        off = (off + bytes + 255) & ~(size_t)255;
        return p;
    };
    half_t* w1h = (half_t*)wsalloc((size_t)N_FC1 * K_FC1 * sizeof(half_t));        // 4.72 MB
    half_t* w2t = (half_t*)wsalloc((size_t)36 * 64 * 64 * sizeof(half_t));         // 0.29 MB
    half_t* a2  = (half_t*)wsalloc((size_t)B_BATCH * 36 * 64 * sizeof(half_t));    // 37.7 MB
    half_t* a1  = (half_t*)wsalloc((size_t)B_BATCH * K_FC1 * sizeof(half_t));      // 37.7 MB
    half_t* hbuf= (half_t*)wsalloc((size_t)B_BATCH * N_FC1 * sizeof(half_t));      // 16.8 MB

    // 1) weight precondition
    prep_weights<<<9792, 256, 0, stream>>>(fc1w, conv2w, w1h, w2t);
    // 2) fused conv1 + relu + patch gather (B*36*64 elements)
    build_p2<<<(B_BATCH * 36 * 64) / 256, 256, 0, stream>>>(x, conv1w, a2);
    // 3) conv2 via WMMA: 256 b-tiles * 36 p * 2 c-tiles = 18432 waves / 8 per block
    conv2_wmma<<<2304, 256, 0, stream>>>(a2, w2t, a1);
    // 4) fc1 via WMMA: 32 b-blocks(256 rows) * 16 n-blocks(64 cols) = 512 blocks
    fc1_wmma<<<512, 256, 0, stream>>>(a1, w1h, hbuf);
    // 5) fc2
    fc2_kernel<<<(B_BATCH * 16) / 256, 256, 0, stream>>>(hbuf, fc2w, out);
}
